// EfficientAttention_90417651515421
// MI455X (gfx1250) — matrix operand order repaired and checked by reading, not compiled, not run on hardware
//
#include <hip/hip_runtime.h>
#include <hip/hip_bf16.h>
#include <math.h>

// Problem constants (from reference)
#define Bdim  4
#define Sdim  4096
#define Edim  1024
#define Hdim  16
#define Ddim  64
#define NSAMP 1024
#define BH    (Bdim * Hdim)   // 64
#define TOK   (Bdim * Sdim)   // 16384

typedef __attribute__((ext_vector_type(16))) __bf16 bf16x16;
typedef __attribute__((ext_vector_type(8)))  float  f32x8;

union Frag32B { bf16x16 v; uint4 q[2]; };

__device__ __forceinline__ unsigned short f2bf(float f) {
  unsigned u = __float_as_uint(f);
  u += 0x7FFFu + ((u >> 16) & 1u);   // round-to-nearest-even
  return (unsigned short)(u >> 16);
}
__device__ __forceinline__ float bf2f(unsigned short h) {
  return __uint_as_float(((unsigned)h) << 16);
}

// ---------------------------------------------------------------------------
// Kernel 1: fused QKV projection GEMM, split-precision bf16 (hi/lo) WMMA.
// Out[t][n] = sum_k x[t][k] * W[n][k] + bias[n], stored as [B,H,S,D].
// WG: 256 thr = 8 waves; tile M=128 (16 rows/wave), N=64, K-step 32.
// ---------------------------------------------------------------------------
#define LDA 40   // padded LDS row stride (elements); 80B rows -> 16B aligned chunks

__global__ __launch_bounds__(256)
void ea_qkv_gemm(const float* __restrict__ x,
                 const float* __restrict__ Wq, const float* __restrict__ bq,
                 const float* __restrict__ Wk, const float* __restrict__ bk,
                 const float* __restrict__ Wv, const float* __restrict__ bv,
                 float* __restrict__ Qo, float* __restrict__ Ko,
                 float* __restrict__ Vo) {
  __shared__ unsigned short AsH[128 * LDA];
  __shared__ unsigned short AsL[128 * LDA];
  __shared__ unsigned short BsH[64 * LDA];
  __shared__ unsigned short BsL[64 * LDA];

  const float* W; const float* bias; float* out;
  if (blockIdx.z == 0)      { W = Wq; bias = bq; out = Qo; }
  else if (blockIdx.z == 1) { W = Wk; bias = bk; out = Ko; }
  else                      { W = Wv; bias = bv; out = Vo; }

  const int tid     = threadIdx.x;
  const int wave    = tid >> 5;
  const int lane    = tid & 31;
  const int laneLow = lane & 15;
  const bool hiHalf = lane >= 16;

  const int t0 = blockIdx.y * 128;   // token base of workgroup tile
  const int n0 = blockIdx.x * 64;    // output-column base (one head per WG tile)
  const int mw = t0 + wave * 16;     // token base of this wave's 16-row strip

  const f32x8 zero8 = {0.f, 0.f, 0.f, 0.f, 0.f, 0.f, 0.f, 0.f};
  f32x8 acc[4];
#pragma unroll
  for (int j = 0; j < 4; ++j) acc[j] = zero8;

  for (int kt = 0; kt < Edim; kt += 32) {
    // ---- stage A (x tile 128x32 f32 -> hi/lo bf16 in LDS) ----
#pragma unroll
    for (int i = 0; i < 4; ++i) {
      int slot = tid + i * 256;          // 0..1023 float4 slots
      int row  = slot >> 3;
      int c4   = (slot & 7) << 2;
      float4 v = *(const float4*)(x + (size_t)(t0 + row) * Edim + kt + c4);
      unsigned short h0 = f2bf(v.x), h1 = f2bf(v.y), h2 = f2bf(v.z), h3 = f2bf(v.w);
      unsigned short l0 = f2bf(v.x - bf2f(h0)), l1 = f2bf(v.y - bf2f(h1));
      unsigned short l2 = f2bf(v.z - bf2f(h2)), l3 = f2bf(v.w - bf2f(h3));
      unsigned long long hp = (unsigned long long)h0 | ((unsigned long long)h1 << 16) |
                              ((unsigned long long)h2 << 32) | ((unsigned long long)h3 << 48);
      unsigned long long lp = (unsigned long long)l0 | ((unsigned long long)l1 << 16) |
                              ((unsigned long long)l2 << 32) | ((unsigned long long)l3 << 48);
      *(unsigned long long*)(AsH + row * LDA + c4) = hp;
      *(unsigned long long*)(AsL + row * LDA + c4) = lp;
    }
    // ---- stage B (W tile 64x32 f32 -> hi/lo bf16 in LDS) ----
#pragma unroll
    for (int i = 0; i < 2; ++i) {
      int slot = tid + i * 256;          // 0..511
      int row  = slot >> 3;
      int c4   = (slot & 7) << 2;
      float4 v = *(const float4*)(W + (size_t)(n0 + row) * Edim + kt + c4);
      unsigned short h0 = f2bf(v.x), h1 = f2bf(v.y), h2 = f2bf(v.z), h3 = f2bf(v.w);
      unsigned short l0 = f2bf(v.x - bf2f(h0)), l1 = f2bf(v.y - bf2f(h1));
      unsigned short l2 = f2bf(v.z - bf2f(h2)), l3 = f2bf(v.w - bf2f(h3));
      unsigned long long hp = (unsigned long long)h0 | ((unsigned long long)h1 << 16) |
                              ((unsigned long long)h2 << 32) | ((unsigned long long)h3 << 48);
      unsigned long long lp = (unsigned long long)l0 | ((unsigned long long)l1 << 16) |
                              ((unsigned long long)l2 << 32) | ((unsigned long long)l3 << 48);
      *(unsigned long long*)(BsH + row * LDA + c4) = hp;
      *(unsigned long long*)(BsL + row * LDA + c4) = lp;
    }
    __syncthreads();

    // ---- A fragments (16x32 bf16: low lanes K0..7|16..23, high lanes K8..15|24..31) ----
    const int abase = hiHalf ? 8 : 0;
    Frag32B aH, aL;
    {
      const unsigned short* ah = AsH + (wave * 16 + laneLow) * LDA;
      const unsigned short* al = AsL + (wave * 16 + laneLow) * LDA;
      aH.q[0] = *(const uint4*)(ah + abase);
      aH.q[1] = *(const uint4*)(ah + abase + 16);
      aL.q[0] = *(const uint4*)(al + abase);
      aL.q[1] = *(const uint4*)(al + abase + 16);
    }
    // ---- B fragments (32x16: lane n = W row, low lanes K0..7|16..23, high K8..15|24..31) ----
    const int bbase = hiHalf ? 8 : 0;
#pragma unroll
    for (int j = 0; j < 4; ++j) {
      const unsigned short* bh = BsH + (j * 16 + laneLow) * LDA + bbase;
      const unsigned short* bl = BsL + (j * 16 + laneLow) * LDA + bbase;
      Frag32B bH, bL;
      bH.q[0] = *(const uint4*)(bh);
      bH.q[1] = *(const uint4*)(bh + 16);
      bL.q[0] = *(const uint4*)(bl);
      bL.q[1] = *(const uint4*)(bl + 16);
      // split-precision: x*W ~= xH*WH + xH*WL + xL*WH  (fp32-class accuracy)
      acc[j] = __builtin_amdgcn_wmma_f32_16x16x32_bf16(false, aH.v, false, bL.v,
                                                       (short)0, acc[j], false, false);
      acc[j] = __builtin_amdgcn_wmma_f32_16x16x32_bf16(false, aL.v, false, bH.v,
                                                       (short)0, acc[j], false, false);
      acc[j] = __builtin_amdgcn_wmma_f32_16x16x32_bf16(false, aH.v, false, bH.v,
                                                       (short)0, acc[j], false, false);
    }
    __syncthreads();
  }

  // ---- epilogue: bias + scatter to [B,H,S,D] ----
#pragma unroll
  for (int j = 0; j < 4; ++j) {
    int ng = n0 + j * 16 + laneLow;
    int h = ng >> 6, d = ng & 63;
    float bb = bias[ng];
#pragma unroll
    for (int r = 0; r < 8; ++r) {
      int t = mw + r + (hiHalf ? 8 : 0);       // C layout: VGPR r -> M=r (+8 high half)
      int b = t >> 12, s = t & (Sdim - 1);
      out[(((size_t)(b * Hdim + h) * Sdim + s) << 6) + d] = acc[j][r] + bb;
    }
  }
}

// ---------------------------------------------------------------------------
// Kernel 2: scores[bh][s] = sum_d Q[bh][s][d]*K[bh][s][d]  (one wave per row)
// ---------------------------------------------------------------------------
__global__ __launch_bounds__(256)
void ea_scores(const float* __restrict__ Q, const float* __restrict__ K,
               float* __restrict__ sc) {
  int row  = (blockIdx.x * blockDim.x + threadIdx.x) >> 5;
  int lane = threadIdx.x & 31;
  if (row >= BH * Sdim) return;
  const float2 q2 = *(const float2*)(Q + (size_t)row * 64 + lane * 2);
  const float2 k2 = *(const float2*)(K + (size_t)row * 64 + lane * 2);
  float p = q2.x * k2.x + q2.y * k2.y;
#pragma unroll
  for (int off = 16; off > 0; off >>= 1) p += __shfl_xor(p, off, 32);
  if (lane == 0) sc[row] = p;
}

// ---------------------------------------------------------------------------
// Kernel 3: per-(b,h) top-1024 of 4096 via MSB-first radix select.
// Tie handling matches jax.lax.top_k (lowest index first among equals).
// ---------------------------------------------------------------------------
__global__ __launch_bounds__(256)
void ea_topk(const float* __restrict__ sc, int* __restrict__ idx) {
  __shared__ unsigned skey[4096];
  __shared__ int red[256];
  __shared__ int scnt;
  const int bh = blockIdx.x, tid = threadIdx.x;
  const float* s = sc + (size_t)bh * Sdim;

  for (int i = tid; i < 4096; i += 256) {
    unsigned u = __float_as_uint(s[i]);
    skey[i] = (u & 0x80000000u) ? ~u : (u | 0x80000000u);  // order-preserving map
  }
  __syncthreads();

  unsigned prefix = 0; int kneed = NSAMP;
  for (int bit = 31; bit >= 0; --bit) {
    unsigned probe = (prefix | (1u << bit)) >> bit;
    int c = 0;
    for (int i = tid; i < 4096; i += 256) c += ((skey[i] >> bit) == probe);
#pragma unroll
    for (int off = 16; off > 0; off >>= 1) c += __shfl_xor(c, off, 32);
    if ((tid & 31) == 0) red[tid >> 5] = c;
    __syncthreads();
    if (tid == 0) { int t = 0; for (int w = 0; w < 8; ++w) t += red[w]; scnt = t; }
    __syncthreads();
    int cnt = scnt;
    if (cnt >= kneed) prefix |= (1u << bit); else kneed -= cnt;
    __syncthreads();
  }
  const unsigned thresh = prefix;   // exact value of 1024th-largest key

  // pass A: strictly greater (order irrelevant; it's a set)
  if (tid == 0) scnt = 0;
  __syncthreads();
  int* outp = idx + (size_t)bh * NSAMP;
  for (int i = tid; i < 4096; i += 256)
    if (skey[i] > thresh) { int p = atomicAdd(&scnt, 1); outp[p] = i; }
  __syncthreads();
  const int gt = scnt;
  const int need_eq = NSAMP - gt;

  // pass B: equals, lowest-index-first via block prefix scan over 16-elem chunks
  const int base = tid * 16;
  int loc = 0;
#pragma unroll
  for (int i = 0; i < 16; ++i) loc += (skey[base + i] == thresh);
  red[tid] = loc;
  __syncthreads();
  for (int off = 1; off < 256; off <<= 1) {
    int v  = red[tid];
    int vo = (tid >= off) ? red[tid - off] : 0;
    __syncthreads();
    red[tid] = v + vo;
    __syncthreads();
  }
  int excl = (tid == 0) ? 0 : red[tid - 1];
  int run = 0;
#pragma unroll
  for (int i = 0; i < 16; ++i) {
    if (skey[base + i] == thresh) {
      int rank = excl + run;
      if (rank < need_eq) outp[gt + rank] = base + i;
      ++run;
    }
  }
}

// ---------------------------------------------------------------------------
// Kernel 4: kv_cos/kv_sin[bh][d][e] = sum_s w(s)*relu(K[bh][s][d]) * V[bh][s][e]
// WG per (b,h); 256 thr = 4 d-groups x 64 e-lanes; 32-row LDS chunks.
// ---------------------------------------------------------------------------
__global__ __launch_bounds__(256)
void ea_kv(const float* __restrict__ K, const float* __restrict__ V,
           float* __restrict__ kvc, float* __restrict__ kvs) {
  __shared__ float kc[32][64], ks[32][64], vv[32][64];
  const int bh = blockIdx.x, tid = threadIdx.x;
  const int e = tid & 63, d0 = (tid >> 6) << 4;
  const float* kp = K + (size_t)bh * Sdim * 64;
  const float* vp = V + (size_t)bh * Sdim * 64;
  float ac[16], as_[16];
#pragma unroll
  for (int j = 0; j < 16; ++j) { ac[j] = 0.f; as_[j] = 0.f; }

  for (int s0 = 0; s0 < Sdim; s0 += 32) {
#pragma unroll
    for (int i = 0; i < 8; ++i) {
      int slot = tid + i * 256;
      int r = slot >> 6, c = slot & 63;
      int sTok = s0 + r;
      float th = 1.5707963267948966f * (float)sTok / (float)Sdim;
      float kr = fmaxf(kp[(size_t)sTok * 64 + c], 0.f);
      kc[r][c] = kr * cosf(th);
      ks[r][c] = kr * sinf(th);
      vv[r][c] = vp[(size_t)sTok * 64 + c];
    }
    __syncthreads();
#pragma unroll 4
    for (int sRow = 0; sRow < 32; ++sRow) {
      float vse = vv[sRow][e];
#pragma unroll
      for (int j = 0; j < 16; ++j) {
        ac[j]  = fmaf(kc[sRow][d0 + j], vse, ac[j]);
        as_[j] = fmaf(ks[sRow][d0 + j], vse, as_[j]);
      }
    }
    __syncthreads();
  }
  float* oc = kvc + (size_t)bh * 4096;
  float* os = kvs + (size_t)bh * 4096;
#pragma unroll
  for (int j = 0; j < 16; ++j) {
    oc[(d0 + j) * 64 + e] = ac[j];
    os[(d0 + j) * 64 + e] = as_[j];
  }
}

// ---------------------------------------------------------------------------
// Kernel 5a: zero the output (float4 stores)
// ---------------------------------------------------------------------------
__global__ __launch_bounds__(256)
void ea_zero(float4* __restrict__ p, int n4) {
  int i = blockIdx.x * 256 + threadIdx.x;
  if (i < n4) p[i] = make_float4(0.f, 0.f, 0.f, 0.f);
}

// ---------------------------------------------------------------------------
// Kernel 5b: gather sampled q, apply feature map, multiply by kv, scatter.
// WG per (b,h): kv (64x64 x2) staged in LDS; 4 token groups x 64 e-lanes.
// Distinct (h,s) targets -> plain stores into zeroed output.
// ---------------------------------------------------------------------------
__global__ __launch_bounds__(256)
void ea_sampled(const float* __restrict__ Q, const int* __restrict__ idx,
                const float* __restrict__ kvc, const float* __restrict__ kvs,
                float* __restrict__ out) {
  __shared__ float c_kv[4096], s_kv[4096];
  __shared__ float qc[4][64], qs[4][64];
  const int bh = blockIdx.x, tid = threadIdx.x;
  const int g = tid >> 6, e = tid & 63;
  const int b = bh >> 4, h = bh & 15;
  for (int i = tid; i < 4096; i += 256) {
    c_kv[i] = kvc[(size_t)bh * 4096 + i];
    s_kv[i] = kvs[(size_t)bh * 4096 + i];
  }
  __syncthreads();
  const float* qp = Q + (size_t)bh * Sdim * 64;
  const int* ip = idx + (size_t)bh * NSAMP;
  for (int n = g; n < NSAMP; n += 4) {   // 256 uniform iterations per group
    int sTok = ip[n];
    float th = 1.5707963267948966f * (float)sTok / (float)Sdim;
    float qr = fmaxf(qp[(size_t)sTok * 64 + e], 0.f);
    qc[g][e] = qr * cosf(th);
    qs[g][e] = qr * sinf(th);
    __syncthreads();
    float acc = 0.f;
#pragma unroll 8
    for (int d = 0; d < 64; ++d) {
      acc = fmaf(qc[g][d], c_kv[d * 64 + e], acc);
      acc = fmaf(qs[g][d], s_kv[d * 64 + e], acc);
    }
    out[((size_t)(b * Sdim + sTok)) * Edim + h * 64 + e] = acc;
    __syncthreads();
  }
}

// ---------------------------------------------------------------------------
extern "C" void kernel_launch(void* const* d_in, const int* in_sizes, int n_in,
                              void* d_out, int out_size, void* d_ws, size_t ws_size,
                              hipStream_t stream) {
  const float* x  = (const float*)d_in[0];
  const float* Wq = (const float*)d_in[1];
  const float* bq = (const float*)d_in[2];
  const float* Wk = (const float*)d_in[3];
  const float* bk = (const float*)d_in[4];
  const float* Wv = (const float*)d_in[5];
  const float* bv = (const float*)d_in[6];
  float* out = (float*)d_out;

  char* w = (char*)d_ws;
  const size_t qkvBytes = (size_t)TOK * Edim * sizeof(float);          // 64 MB each
  float* Q   = (float*)(w);
  float* K   = (float*)(w + qkvBytes);
  float* V   = (float*)(w + 2 * qkvBytes);
  float* sc  = (float*)(w + 3 * qkvBytes);                              // 1 MB
  int*   idx = (int*)  (w + 3 * qkvBytes + (size_t)BH * Sdim * 4);      // 256 KB
  float* kvc = (float*)(w + 3 * qkvBytes + (size_t)BH * Sdim * 4 +
                        (size_t)BH * NSAMP * 4);                        // 1 MB
  float* kvs = kvc + (size_t)BH * Ddim * Ddim;                          // 1 MB

  dim3 gGemm(Edim / 64, TOK / 128, 3);
  ea_qkv_gemm<<<gGemm, 256, 0, stream>>>(x, Wq, bq, Wk, bk, Wv, bv, Q, K, V);

  ea_scores<<<(BH * Sdim) / 8, 256, 0, stream>>>(Q, K, sc);
  ea_topk<<<BH, 256, 0, stream>>>(sc, idx);
  ea_kv<<<BH, 256, 0, stream>>>(K, V, kvc, kvs);
  ea_zero<<<(TOK * Edim / 4) / 256, 256, 0, stream>>>((float4*)out, TOK * Edim / 4);
  ea_sampled<<<BH, 256, 0, stream>>>(Q, idx, kvc, kvs, out);
}
